// GCRNN_38929583571268
// MI455X (gfx1250) — matrix operand order repaired and checked by reading, not compiled, not run on hardware
//
#include <hip/hip_runtime.h>
#include <hip/hip_bf16.h>

#define N_ENT 250002
#define D 128
#define T_STEPS 16
#define E_EDGES 300000
#define S_SEEDS 256
#define COMP_BASE 200000
#define COMP_NUM 50000
#define T_WIN 5
#define NEGF (-1000000000.0f)
#define TS_ROWS (T_STEPS * S_SEEDS)     /* 4096 */
#define NCB ((COMP_NUM + 63) / 64)      /* 782  */

typedef __attribute__((ext_vector_type(16))) __bf16 v16bf;
typedef __attribute__((ext_vector_type(8)))  float  v8f;

union Frag { v16bf v; uint4 u[2]; };

__device__ __forceinline__ float wave_sum(float v) {
#pragma unroll
  for (int o = 16; o > 0; o >>= 1) v += __shfl_xor(v, o, 32);
  return v;
}
__device__ __forceinline__ float wave_max(float v) {
#pragma unroll
  for (int o = 16; o > 0; o >>= 1) v = fmaxf(v, __shfl_xor(v, o, 32));
  return v;
}

// CDNA5 direct global->LDS async copy (16B per lane), tracked by ASYNCcnt.
// lds_off: byte offset in LDS (low 32 bits of a generic shared pointer ==
// addrspacecast flat->local == LDS offset).
__device__ __forceinline__ void async_copy_b128(unsigned lds_off, const void* gsrc) {
  asm volatile("global_load_async_to_lds_b128 %0, %1, off"
               :
               : "v"(lds_off), "v"(gsrc)
               : "memory");
}
__device__ __forceinline__ void wait_asynccnt0() {
  asm volatile("s_wait_asynccnt 0" ::: "memory");
}

// ---------------- message passing ----------------

__global__ __launch_bounds__(256) void k_zero(float4* __restrict__ sums4,
                                              float* __restrict__ cnt) {
  size_t i = (size_t)blockIdx.x * 256 + threadIdx.x;
  if (i < (size_t)N_ENT * 32) sums4[i] = make_float4(0.f, 0.f, 0.f, 0.f);
  if (i < (size_t)N_ENT) cnt[i] = 0.f;
}

// one wave per edge: lanes cover 128 dims as float4; fp32 atomics into L2-resident sums
__global__ __launch_bounds__(256) void k_scatter(const float* __restrict__ node,
                                                 const float* __restrict__ relE,
                                                 const int* __restrict__ src,
                                                 const int* __restrict__ dst,
                                                 const int* __restrict__ rel,
                                                 float* __restrict__ sums,
                                                 float* __restrict__ cnt) {
  __shared__ __align__(16) float relbuf[12 * 128];
  for (int i = threadIdx.x; i < 12 * 128; i += 256) relbuf[i] = relE[i];
  __syncthreads();
  int e    = (blockIdx.x * 256 + threadIdx.x) >> 5;
  int lane = threadIdx.x & 31;
  if (e >= E_EDGES) return;
  int s = src[e], d = dst[e], r = rel[e];
  float4 nv = ((const float4*)(node + (size_t)s * D))[lane];
  float4 rv = ((const float4*)(relbuf + r * D))[lane];
  float4 m  = make_float4(nv.x * rv.x, nv.y * rv.y, nv.z * rv.z, nv.w * rv.w);
  float* sp = sums + (size_t)d * D + lane * 4;
  atomicAdd(sp + 0, m.x); atomicAdd(sp + 1, m.y);
  atomicAdd(sp + 2, m.z); atomicAdd(sp + 3, m.w);
  if (lane == 0) atomicAdd(cnt + d, 1.0f);
}

__global__ __launch_bounds__(256) void k_update(float4* __restrict__ node4,
                                                const float4* __restrict__ sums4,
                                                const float* __restrict__ cnt) {
  size_t i = (size_t)blockIdx.x * 256 + threadIdx.x;
  if (i >= (size_t)N_ENT * 32) return;
  float c = cnt[i >> 5];
  if (c > 0.f) {
    float  inv = 1.0f / c;
    float4 s = sums4[i];
    float4 n = node4[i];
    n.x += s.x * inv; n.y += s.y * inv; n.z += s.z * inv; n.w += s.w * inv;
    node4[i] = n;
  }
}

// after step tau, gather snapshot into stack cache for steps t = tau..tau+4 (w = 4-(t-tau))
__global__ __launch_bounds__(256) void k_gather(const float* __restrict__ node,
                                                const int* __restrict__ seeds,
                                                float* __restrict__ G,
                                                int tau, int nt) {
  int wid  = (blockIdx.x * 256 + threadIdx.x) >> 5;
  int lane = threadIdx.x & 31;
  if (wid >= nt * S_SEEDS) return;
  int j = wid >> 8;            // which future step
  int s = wid & (S_SEEDS - 1);
  int t = tau + j;
  int w = 4 - j;
  int sd = seeds[t * S_SEEDS + s];
  float4 v = ((const float4*)(node + (size_t)sd * D))[lane];
  ((float4*)(G + ((size_t)(t * T_WIN + w) * S_SEEDS + s) * D))[lane] = v;
}

// 5-window attention for step t; one wave per seed
__global__ __launch_bounds__(256) void k_attn(const float* __restrict__ G,
                                              const float* __restrict__ query,
                                              const int* __restrict__ seeds,
                                              float* __restrict__ u,
                                              __hip_bfloat16* __restrict__ ubf,
                                              int t) {
  int wid  = (blockIdx.x * 256 + threadIdx.x) >> 5;
  int lane = threadIdx.x & 31;
  if (wid >= S_SEEDS) return;
  int s  = wid;
  int sd = seeds[t * S_SEEDS + s];
  float4 q = ((const float4*)(query + (size_t)sd * D))[lane];
  int wmin = (t >= T_WIN - 1) ? 0 : (T_WIN - 1 - t);
  float att[T_WIN];
  float amax = -INFINITY;
#pragma unroll
  for (int w = 0; w < T_WIN; ++w) {
    float a = NEGF;
    if (w >= wmin) {
      float4 g = ((const float4*)(G + ((size_t)(t * T_WIN + w) * S_SEEDS + s) * D))[lane];
      a = wave_sum(g.x * q.x + g.y * q.y + g.z * q.z + g.w * q.w);
    }
    att[w] = a;
    amax = fmaxf(amax, a);
  }
  float den = 0.f, wgt[T_WIN];
#pragma unroll
  for (int w = 0; w < T_WIN; ++w) {
    wgt[w] = (w >= wmin) ? expf(att[w] - amax) : 0.f;
    den += wgt[w];
  }
  float4 o = make_float4(0.f, 0.f, 0.f, 0.f);
#pragma unroll
  for (int w = 0; w < T_WIN; ++w) {
    if (w < wmin) continue;
    float ww = wgt[w] / den;
    float4 g = ((const float4*)(G + ((size_t)(t * T_WIN + w) * S_SEEDS + s) * D))[lane];
    o.x += ww * g.x; o.y += ww * g.y; o.z += ww * g.z; o.w += ww * g.w;
  }
  int row = t * S_SEEDS + s;
  ((float4*)(u + (size_t)row * D))[lane] = o;
  __hip_bfloat16* bp = ubf + (size_t)row * D + lane * 4;
  bp[0] = __float2bfloat16(o.x); bp[1] = __float2bfloat16(o.y);
  bp[2] = __float2bfloat16(o.z); bp[3] = __float2bfloat16(o.w);
}

// ---------------- scores GEMM (WMMA bf16) + fused partial logsumexp ----------------

__global__ __launch_bounds__(256) void k_cvt_allc(const float* __restrict__ ent,
                                                  __hip_bfloat16* __restrict__ cbf) {
  size_t i = (size_t)blockIdx.x * 256 + threadIdx.x;  // over COMP_NUM*32 float4s
  if (i >= (size_t)COMP_NUM * 32) return;
  float4 v = ((const float4*)(ent + (size_t)COMP_BASE * D))[i];
  __hip_bfloat16* p = cbf + i * 4;
  p[0] = __float2bfloat16(v.x); p[1] = __float2bfloat16(v.y);
  p[2] = __float2bfloat16(v.z); p[3] = __float2bfloat16(v.w);
}

__global__ void k_init_out(float* out) { if (threadIdx.x == 0) out[0] = 0.f; }

// block tile: 64 rows x 64 cols, K = 128. 8 waves: 4 (M) x 2 (N), each wave 16x32 output.
// Tiles staged via CDNA5 async global->LDS copies (no VGPR round-trip).
__global__ __launch_bounds__(256) void k_gemm_lse(const __hip_bfloat16* __restrict__ A, // [4096][128]
                                                  const __hip_bfloat16* __restrict__ B, // [50000][128]
                                                  float* __restrict__ pmax,
                                                  float* __restrict__ psum) {
  __shared__ __align__(16) __hip_bfloat16 Abuf[64 * 128];
  __shared__ __align__(16) __hip_bfloat16 Bbuf[64 * 128];
  __shared__ __align__(16) float Sbuf[64 * 65];

  int tid     = threadIdx.x;
  int rowBase = blockIdx.x * 64;
  int colBase = blockIdx.y * 64;

  // Async-stage A (64 contiguous rows) and B (64 columns, address-clamped for OOB;
  // garbage scores in OOB columns are skipped by the masked reduction below).
  {
    unsigned ldsA = (unsigned)(uintptr_t)&Abuf[0];
    unsigned ldsB = (unsigned)(uintptr_t)&Bbuf[0];
    const char* Abase = (const char*)(A + (size_t)rowBase * D);
    const char* Bbase = (const char*)B;
#pragma unroll
    for (int c = 0; c < 4; ++c) {
      int idx = tid + c * 256;                   // uint4 index 0..1023
      async_copy_b128(ldsA + (unsigned)idx * 16u, Abase + (size_t)idx * 16);
      int r = idx >> 4, q = idx & 15;
      int col = colBase + r;
      col = (col < COMP_NUM) ? col : (COMP_NUM - 1);
      async_copy_b128(ldsB + (unsigned)idx * 16u,
                      Bbase + (size_t)col * (D * 2) + (size_t)q * 16);
    }
    wait_asynccnt0();   // this wave's LDS writes complete
  }
  __syncthreads();       // all waves' writes complete -> visible block-wide

  int lane = tid & 31;
  int wave = tid >> 5;
  int wm   = (wave & 3) * 16;   // M offset in tile
  int wn   = (wave >> 2) * 32;  // N offset in tile

  v8f acc0 = {}; v8f acc1 = {};
  int ar  = wm + (lane & 15);
  int ak  = (lane & 16) ? 8 : 0;       // A: lanes 0-15 -> K{0..7,16..23}; 16-31 -> K{8..15,24..31}
  int bc0 = wn + (lane & 15);
  int bc1 = bc0 + 16;
  int bk  = (lane & 16) ? 8 : 0;       // B: lanes 0-15 -> K{0..7,16..23}; 16-31 -> K{8..15,24..31}

#pragma unroll
  for (int kk = 0; kk < 4; ++kk) {
    int k0 = kk * 32;
    Frag fa, fb0, fb1;
    fa.u[0]  = *(const uint4*)(Abuf + ar  * D + k0 + ak);
    fa.u[1]  = *(const uint4*)(Abuf + ar  * D + k0 + ak + 16);
    fb0.u[0] = *(const uint4*)(Bbuf + bc0 * D + k0 + bk);
    fb0.u[1] = *(const uint4*)(Bbuf + bc0 * D + k0 + bk + 16);
    fb1.u[0] = *(const uint4*)(Bbuf + bc1 * D + k0 + bk);
    fb1.u[1] = *(const uint4*)(Bbuf + bc1 * D + k0 + bk + 16);
    acc0 = __builtin_amdgcn_wmma_f32_16x16x32_bf16(false, fa.v, false, fb0.v,
                                                   (short)0, acc0, false, false);
    acc1 = __builtin_amdgcn_wmma_f32_16x16x32_bf16(false, fa.v, false, fb1.v,
                                                   (short)0, acc1, false, false);
  }

  // spill scores tile to LDS (C layout: VGPR r -> M = wm + r + (lane>=16 ? 8:0), N = lane&15)
  int mrow = wm + ((lane & 16) ? 8 : 0);
  int ncol = lane & 15;
#pragma unroll
  for (int r = 0; r < 8; ++r) {
    Sbuf[(mrow + r) * 65 + wn + ncol]      = acc0[r];
    Sbuf[(mrow + r) * 65 + wn + 16 + ncol] = acc1[r];
  }
  __syncthreads();

  if (tid < 64) {
    int nvalid = COMP_NUM - colBase;
    if (nvalid > 64) nvalid = 64;
    float vmax = -INFINITY;
    for (int j = 0; j < nvalid; ++j) vmax = fmaxf(vmax, Sbuf[tid * 65 + j]);
    float vsum = 0.f;
    for (int j = 0; j < nvalid; ++j) vsum += expf(Sbuf[tid * 65 + j] - vmax);
    size_t row = (size_t)(rowBase + tid);
    pmax[row * NCB + blockIdx.y] = vmax;
    psum[row * NCB + blockIdx.y] = vsum;
  }
}

// combine per-block partials -> logsumexp, compute pos (fp32), accumulate loss
__global__ __launch_bounds__(256) void k_lse_loss(const float* __restrict__ pmax,
                                                  const float* __restrict__ psum,
                                                  const float* __restrict__ u,
                                                  const float* __restrict__ ent,
                                                  const int* __restrict__ cidx,
                                                  float* __restrict__ out) {
  int wave = (blockIdx.x * 256 + threadIdx.x) >> 5;
  int lane = threadIdx.x & 31;
  if (wave >= TS_ROWS) return;
  float m = -INFINITY;
  for (int i = lane; i < NCB; i += 32) m = fmaxf(m, pmax[(size_t)wave * NCB + i]);
  m = wave_max(m);
  float s = 0.f;
  for (int i = lane; i < NCB; i += 32)
    s += psum[(size_t)wave * NCB + i] * expf(pmax[(size_t)wave * NCB + i] - m);
  s = wave_sum(s);
  float lse = m + logf(s);
  const float* urow = u + (size_t)wave * D;
  const float* trow = ent + (size_t)(COMP_BASE + cidx[wave]) * D;
  float p = 0.f;
  for (int d = lane; d < D; d += 32) p += urow[d] * trow[d];
  p = wave_sum(p);
  if (lane == 0) atomicAdd(out, -(p - lse));
}

// ---------------- host orchestration ----------------

extern "C" void kernel_launch(void* const* d_in, const int* in_sizes, int n_in,
                              void* d_out, int out_size, void* d_ws, size_t ws_size,
                              hipStream_t stream) {
  (void)in_sizes; (void)n_in; (void)out_size; (void)ws_size;
  const float* ent   = (const float*)d_in[0];
  const float* query = (const float*)d_in[1];
  const float* relE  = (const float*)d_in[2];
  const int*   esrc  = (const int*)d_in[3];
  const int*   edst  = (const int*)d_in[4];
  const int*   erel  = (const int*)d_in[5];
  const int*   seeds = (const int*)d_in[6];
  const int*   cidx  = (const int*)d_in[7];
  float* out = (float*)d_out;

  char* ws = (char*)d_ws;
  size_t off = 0;
  auto alloc = [&](size_t bytes) -> void* {
    void* p = ws + off;
    off += (bytes + 255) & ~(size_t)255;
    return p;
  };
  float* node = (float*)alloc((size_t)N_ENT * D * 4);
  float* sums = (float*)alloc((size_t)N_ENT * D * 4);
  float* cnt  = (float*)alloc((size_t)N_ENT * 4);
  float* G    = (float*)alloc((size_t)T_STEPS * T_WIN * S_SEEDS * D * 4);
  float* u    = (float*)alloc((size_t)TS_ROWS * D * 4);
  __hip_bfloat16* ubf = (__hip_bfloat16*)alloc((size_t)TS_ROWS * D * 2);
  __hip_bfloat16* cbf = (__hip_bfloat16*)alloc((size_t)COMP_NUM * D * 2);
  float* pmax = (float*)alloc((size_t)TS_ROWS * NCB * 4);
  float* psum = (float*)alloc((size_t)TS_ROWS * NCB * 4);

  hipMemcpyAsync(node, ent, (size_t)N_ENT * D * 4, hipMemcpyDeviceToDevice, stream);

  int zgrid = (int)(((size_t)N_ENT * 32 + 255) / 256);
  for (int t = 0; t < T_STEPS; ++t) {
    k_zero<<<zgrid, 256, 0, stream>>>((float4*)sums, cnt);
    k_scatter<<<(E_EDGES + 7) / 8, 256, 0, stream>>>(
        node, relE, esrc + (size_t)t * E_EDGES, edst + (size_t)t * E_EDGES,
        erel + (size_t)t * E_EDGES, sums, cnt);
    k_update<<<zgrid, 256, 0, stream>>>((float4*)node, (const float4*)sums, cnt);
    int nt = (T_STEPS - t) < T_WIN ? (T_STEPS - t) : T_WIN;
    k_gather<<<nt * 32, 256, 0, stream>>>(node, seeds, G, t, nt);
    k_attn<<<32, 256, 0, stream>>>(G, query, seeds, u, ubf, t);
  }

  k_cvt_allc<<<(COMP_NUM * 32 + 255) / 256, 256, 0, stream>>>(ent, cbf);
  k_init_out<<<1, 32, 0, stream>>>(out);
  dim3 gg(TS_ROWS / 64, NCB);
  k_gemm_lse<<<gg, 256, 0, stream>>>(ubf, cbf, pmax, psum);
  k_lse_loss<<<TS_ROWS / 8, 256, 0, stream>>>(pmax, psum, u, ent, cidx, out);
}
